// WaveNet_28776280883514
// MI455X (gfx1250) — matrix operand order repaired and checked by reading, not compiled, not run on hardware
//
#include <hip/hip_runtime.h>
#include <hip/hip_bf16.h>

typedef __attribute__((ext_vector_type(16))) _Float16 v16h;
typedef __attribute__((ext_vector_type(8)))  _Float16 v8h;
typedef __attribute__((ext_vector_type(8)))  float    v8f;

__device__ __forceinline__ _Float16 relu16(_Float16 v) {
    return (v > (_Float16)0) ? v : (_Float16)0;
}

// ---------------- embedding gather: int tokens -> f16 activations ----------
__global__ void wn_embed_kernel(const int* __restrict__ x,
                                const float* __restrict__ emb,
                                _Float16* __restrict__ h, int T) {
    int i = blockIdx.x * blockDim.x + threadIdx.x;
    int total = 4 * T * 16;
    if (i >= total) return;
    int ci  = i & 15;
    int bt  = i >> 4;
    int pos = ci >> 3;
    int c   = ci & 7;
    int tok = x[bt * 2 + pos];
    h[i] = (_Float16)emb[tok * 8 + c];
}

// ---- repack f32 weights [K x W] into f16 WMMA B-fragment layout -----------
// dst[((kc*(W/16)+nt)*32 + lane)*16 + e]  where K(e)=kc*32+8*half+(e&7)+16*(e>>3),
// N=nt*16+lrow.  Rows kk >= Kvalid are zero (K padding for GEMM2).
__global__ void wn_repack_kernel(const float* __restrict__ src,
                                 _Float16* __restrict__ dst,
                                 int W, int Kvalid, int Kpad) {
    int i = blockIdx.x * blockDim.x + threadIdx.x;
    int total = Kpad * W;
    if (i >= total) return;
    int e    = i & 15;
    int lane = (i >> 4) & 31;
    int tile = i >> 9;                 // kc*(W/16) + nt
    int nts  = W >> 4;
    int kc   = tile / nts;
    int nt   = tile - kc * nts;
    int half = lane >> 4, lrow = lane & 15;
    int kk   = kc * 32 + half * 8 + (e & 7) + ((e >> 3) << 4);
    int co   = nt * 16 + lrow;
    dst[i] = (kk < Kvalid) ? (_Float16)src[kk * W + co] : (_Float16)0;
}

// ---------------- one WaveNet residual layer, fully fused ------------------
template <int CI, int W>
__global__ __launch_bounds__(128) void wn_layer_kernel(
    const _Float16* __restrict__ hin, _Float16* __restrict__ hout,
    const _Float16* __restrict__ wp1, const float* __restrict__ b1,
    const _Float16* __restrict__ wp2, const float* __restrict__ b2,
    const float* __restrict__ we, const float* __restrict__ be,
    int Tin, int Tout, int dil, int tiles) {

    constexpr int K1   = 2 * CI;
    constexpr int H    = W / 2;
    constexpr int K2   = (H < 32) ? 32 : H;
    constexpr int NT   = W / 16;
    constexpr int CLOG = (CI == 16) ? 4 : (CI == 32) ? 5 : (CI == 64) ? 6 : 7;
    constexpr int MT   = 64;
    constexpr int CV   = CI / 8;            // v8h vectors per activation row

    __shared__ _Float16 sb[2][MT][CI];      // raw taps (tap1 = residual x0)
    __shared__ _Float16 hb[MT][W];          // h1 = conv1 + b1
    __shared__ _Float16 zb[MT][K2];         // gated, zero padded

    const int tid  = threadIdx.x;
    const int wid  = tid >> 5;
    const int lane = tid & 31;
    const int half = lane >> 4;
    const int lrow = lane & 15;
    const int mrow = wid * 16 + lrow;

    const int bb = blockIdx.x / tiles;
    const int tb = (blockIdx.x - bb * tiles) * MT;
    const bool fullTile = (tb + MT <= Tout);

    // ---- stage both tap windows (raw, pre-ReLU) into LDS; 16B vectors ----
    for (int i = tid; i < 2 * MT * CV; i += 128) {
        int tap = i / (MT * CV);
        int rem = i - tap * (MT * CV);
        int m   = rem / CV;
        int cv  = rem - m * CV;
        int t   = tb + m + tap * dil;
        v8h v = {};
        if (t < Tin)
            v = *(const v8h*)&hin[(size_t)(bb * Tin + t) * CI + cv * 8];
        *(v8h*)&sb[tap][m][cv * 8] = v;
    }
    __syncthreads();

    // ---- GEMM1: h1 = relu(x) (*) w1 (taps folded into K) ----
    {
        v8f acc[NT];
#pragma unroll
        for (int nt = 0; nt < NT; ++nt) acc[nt] = {};
        for (int kc = 0; kc < K1 / 32; ++kc) {
            const int kkA = kc * 32 + half * 8;
            const int kkB = kkA + 16;
            const v8h r0 = *(const v8h*)&sb[kkA >> CLOG][mrow][kkA & (CI - 1)];
            const v8h r1 = *(const v8h*)&sb[kkB >> CLOG][mrow][kkB & (CI - 1)];
            v16h a;
#pragma unroll
            for (int e = 0; e < 8; ++e) {
                a[e]     = relu16(r0[e]);
                a[e + 8] = relu16(r1[e]);
            }
            __builtin_prefetch(&wp1[(size_t)((kc + 1) * NT) * 512], 0, 1);
#pragma unroll
            for (int nt = 0; nt < NT; ++nt) {
                const v16h bf =
                    *(const v16h*)&wp1[((size_t)(kc * NT + nt) * 32 + lane) * 16];
                acc[nt] = __builtin_amdgcn_wmma_f32_16x16x32_f16(
                    false, a, false, bf, (short)0, acc[nt], false, false);
            }
        }
#pragma unroll
        for (int nt = 0; nt < NT; ++nt) {
            const int co = nt * 16 + lrow;
            float bias = b1[co];
#pragma unroll
            for (int r = 0; r < 8; ++r)       // D layout: M = r + 8*half
                hb[wid * 16 + r + half * 8][co] = (_Float16)(acc[nt][r] + bias);
        }
    }
    __syncthreads();

    // ---- gate: z = tanh(filt) * sigmoid(gate) ----
    for (int i = tid; i < MT * H; i += 128) {
        int m = i / H, c = i - m * H;
        float f = (float)hb[m][c];
        float g = (float)hb[m][c + H];
        zb[m][c] = (_Float16)(tanhf(f) / (1.0f + __expf(-g)));
    }
    if constexpr (K2 > H) {
        for (int i = tid; i < MT * (K2 - H); i += 128) {
            int m = i / (K2 - H), c = H + (i - m * (K2 - H));
            zb[m][c] = (_Float16)0;
        }
    }
    __syncthreads();

    // ---- GEMM2 (1x1 conv) + bias + residual epilogue ----
    {
        v8f acc[NT];
#pragma unroll
        for (int nt = 0; nt < NT; ++nt) acc[nt] = {};
#pragma unroll
        for (int kc = 0; kc < K2 / 32; ++kc) {
            const int kkA = kc * 32 + half * 8;
            const v8h r0 = *(const v8h*)&zb[mrow][kkA];
            const v8h r1 = *(const v8h*)&zb[mrow][kkA + 16];
            v16h a;
#pragma unroll
            for (int e = 0; e < 8; ++e) {
                a[e]     = r0[e];
                a[e + 8] = r1[e];
            }
#pragma unroll
            for (int nt = 0; nt < NT; ++nt) {
                const v16h bf =
                    *(const v16h*)&wp2[((size_t)(kc * NT + nt) * 32 + lane) * 16];
                acc[nt] = __builtin_amdgcn_wmma_f32_16x16x32_f16(
                    false, a, false, bf, (short)0, acc[nt], false, false);
            }
        }

        // epilogue: guard==false is the block-uniform full-tile fast path
        auto epilogue = [&](bool guard) {
#pragma unroll
            for (int nt = 0; nt < NT; ++nt) {
                const int co = nt * 16 + lrow;
                float bias = b2[co];
#pragma unroll
                for (int r = 0; r < 8; ++r) {
                    int ml = wid * 16 + r + half * 8;
                    int t  = tb + ml;
                    if (!guard || t < Tout) {
                        float res;
                        if constexpr (W > CI) {
                            if (co < CI) {
                                res = (float)sb[1][ml][co];
                            } else {   // channel-expansion residual (we/be)
                                float s = be[co - CI];
                                for (int ci = 0; ci < CI; ++ci)
                                    s += (float)sb[1][ml][ci] *
                                         we[ci * (W - CI) + (co - CI)];
                                res = s;
                            }
                        } else {
                            res = (float)sb[1][ml][co];
                        }
                        hout[(size_t)(bb * Tout + t) * W + co] =
                            (_Float16)(acc[nt][r] + bias + res);
                    }
                }
            }
        };
        if (fullTile) epilogue(false);
        else          epilogue(true);
    }
}

// ---------------- final: relu -> 1x1 conv 128 -> 512, f32 out --------------
__global__ __launch_bounds__(128) void wn_final_kernel(
    const _Float16* __restrict__ hin, float* __restrict__ out,
    const _Float16* __restrict__ wpF, const float* __restrict__ ob,
    int T, int tiles) {

    __shared__ _Float16 sb[64][128];

    const int tid  = threadIdx.x;
    const int wid  = tid >> 5;
    const int lane = tid & 31;
    const int half = lane >> 4;
    const int lrow = lane & 15;
    const int mrow = wid * 16 + lrow;

    const int bb = blockIdx.x / tiles;
    const int tb = (blockIdx.x % tiles) * 64;
    const bool fullTile = (tb + 64 <= T);

    for (int i = tid; i < 64 * 16; i += 128) {   // 16 x v8h per row
        int m  = i >> 4;
        int cv = i & 15;
        int t  = tb + m;
        v8h v = {};
        if (t < T)
            v = *(const v8h*)&hin[(size_t)(bb * T + t) * 128 + cv * 8];
#pragma unroll
        for (int e = 0; e < 8; ++e) v[e] = relu16(v[e]);
        *(v8h*)&sb[m][cv * 8] = v;
    }
    __syncthreads();

    const int NTF = 32;                    // 512 / 16 output tiles
    for (int nt = 0; nt < NTF; ++nt) {
        v8f acc = {};
        const int co = nt * 16 + lrow;
#pragma unroll
        for (int kc = 0; kc < 4; ++kc) {
            const int kkA = kc * 32 + half * 8;
            const v8h r0 = *(const v8h*)&sb[mrow][kkA];
            const v8h r1 = *(const v8h*)&sb[mrow][kkA + 16];
            v16h a;
#pragma unroll
            for (int e = 0; e < 8; ++e) {
                a[e]     = r0[e];
                a[e + 8] = r1[e];
            }
            const v16h bf =
                *(const v16h*)&wpF[((size_t)(kc * NTF + nt) * 32 + lane) * 16];
            acc = __builtin_amdgcn_wmma_f32_16x16x32_f16(
                false, a, false, bf, (short)0, acc, false, false);
        }
        float bias = ob[co];
        if (fullTile) {
#pragma unroll
            for (int r = 0; r < 8; ++r) {
                int t = tb + wid * 16 + r + half * 8;
                out[(size_t)(bb * T + t) * 512 + co] = acc[r] + bias;
            }
        } else {
#pragma unroll
            for (int r = 0; r < 8; ++r) {
                int t = tb + wid * 16 + r + half * 8;
                if (t < T)
                    out[(size_t)(bb * T + t) * 512 + co] = acc[r] + bias;
            }
        }
    }
}

// ---------------------------------------------------------------------------
extern "C" void kernel_launch(void* const* d_in, const int* in_sizes, int n_in,
                              void* d_out, int out_size, void* d_ws, size_t ws_size,
                              hipStream_t stream) {
    const int B  = 4;
    const int T0 = 44099;

    const int*   x   = (const int*)d_in[0];
    const float* emb = (const float*)d_in[1];

    // workspace layout: two f16 ping-pong activation buffers + f16 weight pool
    const size_t bufElems = 10000000;          // >= max per-layer 4*T*C
    _Float16* bufA = (_Float16*)d_ws;
    _Float16* bufB = bufA + bufElems;
    _Float16* wpool = bufB + bufElems;         // repacked-weight cursor
    size_t wcur = 0;

    {
        int total = B * T0 * 16;
        wn_embed_kernel<<<(total + 255) / 256, 256, 0, stream>>>(x, emb, bufA, T0);
    }

    _Float16* cur = bufA;
    _Float16* nxt = bufB;
    int idx = 2, in_ch = 16, Tcur = T0;

    for (int blk = 0; blk < 4; ++blk) {
        int width = 16 << blk;
        for (int o = 0; o < 13; ++o) {
            int dil = 1 << o;
            const float* w1 = (const float*)d_in[idx++];
            const float* b1 = (const float*)d_in[idx++];
            const float* w2 = (const float*)d_in[idx++];
            const float* b2 = (const float*)d_in[idx++];
            const float* we = nullptr;
            const float* be = nullptr;
            if (width > in_ch) {
                we = (const float*)d_in[idx++];
                be = (const float*)d_in[idx++];
            }
            int K1 = 2 * in_ch;
            int H  = width / 2;
            int K2 = (H < 32) ? 32 : H;

            _Float16* wp1 = wpool + wcur; wcur += (size_t)K1 * width;
            _Float16* wp2 = wpool + wcur; wcur += (size_t)K2 * width;
            {
                int t1 = K1 * width;
                wn_repack_kernel<<<(t1 + 255) / 256, 256, 0, stream>>>(
                    w1, wp1, width, K1, K1);
                int t2 = K2 * width;
                wn_repack_kernel<<<(t2 + 255) / 256, 256, 0, stream>>>(
                    w2, wp2, width, H, K2);
            }

            int Tout  = Tcur - dil;
            int tiles = (Tout + 63) / 64;
            dim3 grid(B * tiles);

#define WN_LAUNCH(CI_, W_)                                                  \
    wn_layer_kernel<CI_, W_><<<grid, 128, 0, stream>>>(                     \
        cur, nxt, wp1, b1, wp2, b2, we, be, Tcur, Tout, dil, tiles)

            if      (in_ch ==  16 && width ==  16) WN_LAUNCH(16, 16);
            else if (in_ch ==  16 && width ==  32) WN_LAUNCH(16, 32);
            else if (in_ch ==  32 && width ==  32) WN_LAUNCH(32, 32);
            else if (in_ch ==  32 && width ==  64) WN_LAUNCH(32, 64);
            else if (in_ch ==  64 && width ==  64) WN_LAUNCH(64, 64);
            else if (in_ch ==  64 && width == 128) WN_LAUNCH(64, 128);
            else if (in_ch == 128 && width == 128) WN_LAUNCH(128, 128);
#undef WN_LAUNCH

            _Float16* t = cur; cur = nxt; nxt = t;
            in_ch = width;
            Tcur  = Tout;
        }
    }

    const float* ow = (const float*)d_in[idx++];
    const float* ob = (const float*)d_in[idx++];

    _Float16* wpF = wpool + wcur; wcur += (size_t)128 * 512;
    {
        int tF = 128 * 512;
        wn_repack_kernel<<<(tF + 255) / 256, 256, 0, stream>>>(
            ow, wpF, 512, 128, 128);
    }

    int tiles = (Tcur + 63) / 64;
    wn_final_kernel<<<dim3(B * tiles), 128, 0, stream>>>(
        cur, (float*)d_out, wpF, ob, Tcur, tiles);
}
